// Connector_84344567759357
// MI455X (gfx1250) — matrix operand order repaired and checked by reading, not compiled, not run on hardware
//
#include <hip/hip_runtime.h>
#include <hip/hip_bf16.h>
#include <math.h>

typedef __attribute__((ext_vector_type(16))) _Float16 v16h;
typedef __attribute__((ext_vector_type(8)))  float    v8f;

#define EPSK 1e-5f

constexpr int N4     = 4;
constexpr int CDIM   = 2048;
constexpr int NC     = 8192;      // N4*CDIM (K of the GEMV)
constexpr int TOK    = 8192;      // B*S
constexpr int KSTEPS = NC / 32;   // 256 WMMA K-steps
constexpr int MT     = 16;        // tokens per block in gates kernel
constexpr int NB1    = TOK / MT;  // 512 blocks

// ---------------------------------------------------------------------------
// Kernel 0: pack phi_{pre,post,res} (f32) into f16 WMMA B-fragments.
// B is 32(K) x 16(N) per tile; per ISA layout: lane = 16*kh + n holds
// halves h = B[k = 8*kh + h + (h >= 8 ? 8 : 0)][n].  Two N-tiles: cols 0-3 = pre, 4-7 = post,
// 8-23 = res, 24-31 = zero pad.
// pack index: (((step*2 + tile)*32 + lane) * 16) halves  -> 512 KB total.
// ---------------------------------------------------------------------------
__global__ void __launch_bounds__(256)
pack_phi_kernel(const float* __restrict__ phi_pre,
                const float* __restrict__ phi_post,
                const float* __restrict__ phi_res,
                _Float16* __restrict__ pack) {
  int g = blockIdx.x * blockDim.x + threadIdx.x;   // 0 .. 16383
  if (g >= KSTEPS * 2 * 32) return;
  int lane = g & 31;
  int tile = (g >> 5) & 1;
  int step = g >> 6;
  int n    = tile * 16 + (lane & 15);
  int kh   = lane >> 4;
  v16h vh;
#pragma unroll
  for (int h = 0; h < 16; ++h) {
    int k = step * 32 + 8 * kh + h + (h >= 8 ? 8 : 0);
    float v = 0.0f;
    if (n < 4)       v = phi_pre[k * 4 + n];
    else if (n < 8)  v = phi_post[k * 4 + (n - 4)];
    else if (n < 24) v = phi_res[k * 16 + (n - 8)];
    vh[h] = (_Float16)v;
  }
  *(v16h*)(pack + (size_t)g * 16) = vh;
}

// ---------------------------------------------------------------------------
// Kernel 1: per-token gates.
//   raw[t][0..23] = x[t] @ phiAll  (WMMA f32_16x16x32_f16, K=8192)
//   ss[t]         = sum x^2  (accumulated during A loads)
//   -> H_pre, H_post, M = sinkhorn(0.1*raw/rms + b_res), 24 floats/token.
// 16 tokens per block, 8 waves each own K-stripe of 1024, LDS reduction.
// ---------------------------------------------------------------------------
__global__ void __launch_bounds__(256)
gates_kernel(const float* __restrict__ x,
             const _Float16* __restrict__ pack,
             const float* __restrict__ b_pre,
             const float* __restrict__ b_post,
             const float* __restrict__ b_res,
             const float* __restrict__ a_pre,
             const float* __restrict__ a_post,
             const float* __restrict__ a_res,
             float* __restrict__ gates) {
  __shared__ float cred[8][2][256];   // per-wave C tiles, [m*16+n]
  __shared__ float rawb[2][256];      // reduced raw, [tile][m*16+n]
  __shared__ float ss_s[16];

  const int tid  = threadIdx.x;
  const int wave = tid >> 5;
  const int lane = tid & 31;
  const int m    = lane & 15;   // token-in-tile (= N index for C store)
  const int kg   = lane >> 4;   // lane half-group
  const int tok0 = blockIdx.x * MT;

  if (tid < 16) ss_s[tid] = 0.0f;
  __syncthreads();

  const float* row = x + (size_t)(tok0 + m) * NC;
  v8f acc0 = {};
  v8f acc1 = {};
  float ss = 0.0f;
  const int kbase = wave * 1024;

#pragma unroll 2
  for (int j = 0; j < 32; ++j) {
    const int k0 = kbase + j * 32;
    // A fragment (16x32 f16): lane kg=0 holds K runs [0..7],[16..23];
    // kg=1 holds [8..15],[24..31] of this 32-wide K slab.
    float4 f0 = *(const float4*)(row + k0 + 8 * kg);
    float4 f1 = *(const float4*)(row + k0 + 8 * kg + 4);
    float4 f2 = *(const float4*)(row + k0 + 16 + 8 * kg);
    float4 f3 = *(const float4*)(row + k0 + 20 + 8 * kg);
    ss += f0.x * f0.x + f0.y * f0.y + f0.z * f0.z + f0.w * f0.w;
    ss += f1.x * f1.x + f1.y * f1.y + f1.z * f1.z + f1.w * f1.w;
    ss += f2.x * f2.x + f2.y * f2.y + f2.z * f2.z + f2.w * f2.w;
    ss += f3.x * f3.x + f3.y * f3.y + f3.z * f3.z + f3.w * f3.w;
    v16h a;
    a[0]  = (_Float16)f0.x; a[1]  = (_Float16)f0.y;
    a[2]  = (_Float16)f0.z; a[3]  = (_Float16)f0.w;
    a[4]  = (_Float16)f1.x; a[5]  = (_Float16)f1.y;
    a[6]  = (_Float16)f1.z; a[7]  = (_Float16)f1.w;
    a[8]  = (_Float16)f2.x; a[9]  = (_Float16)f2.y;
    a[10] = (_Float16)f2.z; a[11] = (_Float16)f2.w;
    a[12] = (_Float16)f3.x; a[13] = (_Float16)f3.y;
    a[14] = (_Float16)f3.z; a[15] = (_Float16)f3.w;

    const int sgl = wave * 32 + j;  // global K-step
    const v16h b0 = *(const v16h*)(pack + ((size_t)(sgl * 2 + 0) * 32 + lane) * 16);
    const v16h b1 = *(const v16h*)(pack + ((size_t)(sgl * 2 + 1) * 32 + lane) * 16);

    acc0 = __builtin_amdgcn_wmma_f32_16x16x32_f16(false, a, false, b0,
                                                  (short)0, acc0, false, false);
    acc1 = __builtin_amdgcn_wmma_f32_16x16x32_f16(false, a, false, b1,
                                                  (short)0, acc1, false, false);
  }

  atomicAdd(&ss_s[m], ss);
  // C layout: VGPR r -> M = r + 8*(lane/16), N = lane & 15
#pragma unroll
  for (int r = 0; r < 8; ++r) {
    const int mm = r + 8 * kg;
    cred[wave][0][mm * 16 + m] = acc0[r];
    cred[wave][1][mm * 16 + m] = acc1[r];
  }
  __syncthreads();

  float r0 = 0.0f, r1 = 0.0f;
#pragma unroll
  for (int w = 0; w < 8; ++w) { r0 += cred[w][0][tid]; r1 += cred[w][1][tid]; }
  rawb[0][tid] = r0;
  rawb[1][tid] = r1;
  __syncthreads();

  if (tid < 16) {
    const float inv = 1.0f / sqrtf(ss_s[tid] * (1.0f / NC) + EPSK);
    const float ap  = a_pre[0]  * inv;
    const float apo = a_post[0] * inv;
    const float ar  = a_res[0]  * inv;
    float* g = gates + (size_t)(tok0 + tid) * 24;
#pragma unroll
    for (int j = 0; j < 4; ++j) {
      const float t = ap * rawb[0][tid * 16 + j] + b_pre[j];
      g[j] = 1.0f / (1.0f + expf(-t));
    }
#pragma unroll
    for (int j = 0; j < 4; ++j) {
      const float t = apo * rawb[0][tid * 16 + 4 + j] + b_post[j];
      g[4 + j] = 2.0f / (1.0f + expf(-t));
    }
    float Mm[16];
#pragma unroll
    for (int e = 0; e < 16; ++e) {
      const int n = 8 + e;  // global GEMV column
      const float raw = (n < 16) ? rawb[0][tid * 16 + n]
                                 : rawb[1][tid * 16 + (n - 16)];
      Mm[e] = expf(ar * raw + b_res[e]);
    }
    for (int it = 0; it < 20; ++it) {
#pragma unroll
      for (int i = 0; i < 4; ++i) {
        const float rs = Mm[i*4] + Mm[i*4+1] + Mm[i*4+2] + Mm[i*4+3] + EPSK;
        const float iv = 1.0f / rs;
        Mm[i*4] *= iv; Mm[i*4+1] *= iv; Mm[i*4+2] *= iv; Mm[i*4+3] *= iv;
      }
#pragma unroll
      for (int jc = 0; jc < 4; ++jc) {
        const float cs = Mm[jc] + Mm[4+jc] + Mm[8+jc] + Mm[12+jc] + EPSK;
        const float iv = 1.0f / cs;
        Mm[jc] *= iv; Mm[4+jc] *= iv; Mm[8+jc] *= iv; Mm[12+jc] *= iv;
      }
    }
#pragma unroll
    for (int e = 0; e < 16; ++e) g[8 + e] = Mm[e];
  }
}

// ---------------------------------------------------------------------------
// Kernel 2: one block per token; x row held in registers, read once.
//   agg = H_pre . x ; rms2 over C ; out = M@x + H_post * (w*agg/rms2)
// ---------------------------------------------------------------------------
__global__ void __launch_bounds__(256)
mix_kernel(const float* __restrict__ x,
           const float* __restrict__ gates,
           const float* __restrict__ nw,
           float* __restrict__ out) {
  __shared__ float g_s[24];
  __shared__ float red[256];
  __shared__ float inv_s;
  const int tid = threadIdx.x;
  const size_t tok = blockIdx.x;

  if (tid < 24) g_s[tid] = gates[tok * 24 + tid];

  const float* xr = x + tok * NC;
  float xv[4][8];
#pragma unroll
  for (int u = 0; u < 8; ++u) {
    const int c = tid + u * 256;
#pragma unroll
    for (int j = 0; j < 4; ++j) xv[j][u] = xr[j * CDIM + c];
  }
  __syncthreads();

  const float hp0 = g_s[0], hp1 = g_s[1], hp2 = g_s[2], hp3 = g_s[3];
  float agg[8];
  float ss = 0.0f;
#pragma unroll
  for (int u = 0; u < 8; ++u) {
    const float a = hp0 * xv[0][u] + hp1 * xv[1][u] + hp2 * xv[2][u] + hp3 * xv[3][u];
    agg[u] = a;
    ss += a * a;
  }
  red[tid] = ss;
  __syncthreads();
  if (tid < 32) {
    float s = 0.0f;
#pragma unroll
    for (int i = 0; i < 8; ++i) s += red[tid + 32 * i];
    red[tid] = s;
  }
  __syncthreads();
  if (tid == 0) {
    float t = 0.0f;
    for (int i = 0; i < 32; ++i) t += red[i];
    inv_s = 1.0f / sqrtf(t * (1.0f / CDIM) + EPSK);
  }
  __syncthreads();

  const float inv = inv_s;
  float Mr[16], hq[4];
#pragma unroll
  for (int e = 0; e < 16; ++e) Mr[e] = g_s[8 + e];
#pragma unroll
  for (int i = 0; i < 4; ++i) hq[i] = g_s[4 + i];

  float* orow = out + tok * NC;
#pragma unroll
  for (int u = 0; u < 8; ++u) {
    const int c = tid + u * 256;
    const float xn = nw[c] * agg[u] * inv;
#pragma unroll
    for (int i = 0; i < 4; ++i) {
      const float v = Mr[i*4+0] * xv[0][u] + Mr[i*4+1] * xv[1][u] +
                      Mr[i*4+2] * xv[2][u] + Mr[i*4+3] * xv[3][u] + hq[i] * xn;
      orow[i * CDIM + c] = v;
    }
  }
}

// ---------------------------------------------------------------------------
extern "C" void kernel_launch(void* const* d_in, const int* in_sizes, int n_in,
                              void* d_out, int out_size, void* d_ws, size_t ws_size,
                              hipStream_t stream) {
  const float* x        = (const float*)d_in[0];
  const float* phi_pre  = (const float*)d_in[1];
  const float* phi_post = (const float*)d_in[2];
  const float* phi_res  = (const float*)d_in[3];
  const float* b_pre    = (const float*)d_in[4];
  const float* b_post   = (const float*)d_in[5];
  const float* b_res    = (const float*)d_in[6];
  const float* a_pre    = (const float*)d_in[7];
  const float* a_post   = (const float*)d_in[8];
  const float* a_res    = (const float*)d_in[9];
  const float* nw       = (const float*)d_in[10];

  // ws layout: [0, 512KB) packed phi f16 fragments; [512KB, +768KB) gates.
  _Float16* pack  = (_Float16*)d_ws;
  float*    gates = (float*)((char*)d_ws + 512 * 1024);
  float*    out   = (float*)d_out;

  pack_phi_kernel<<<64, 256, 0, stream>>>(phi_pre, phi_post, phi_res, pack);
  gates_kernel<<<NB1, 256, 0, stream>>>(x, pack, b_pre, b_post, b_res,
                                        a_pre, a_post, a_res, gates);
  mix_kernel<<<TOK, 256, 0, stream>>>(x, gates, nw, out);
}
